// Encoder_48112223650530
// MI455X (gfx1250) — matrix operand order repaired and checked by reading, not compiled, not run on hardware
//
#include <hip/hip_runtime.h>

// MI455X / gfx1250 encoder: bf16 WMMA (f32 accumulate) GEMMs + flash attention
// with CDNA5 async global->LDS staging (double-buffered, ASYNCcnt-tracked) and
// band-sparse relative-position terms (|s-t| <= WS).

typedef __attribute__((ext_vector_type(16))) __bf16 v16bf;
typedef __attribute__((ext_vector_type(8)))  float  v8f;

constexpr int B_  = 4;
constexpr int D_  = 192;
constexpr int T_  = 2048;
constexpr int H_  = 2;
constexpr int KC_ = 96;
constexpr int FF_ = 768;
constexpr int L_  = 6;
constexpr int KS_ = 3;
constexpr int WS_ = 10;
constexpr int NR_ = 2 * WS_ + 1;   // 21 relative offsets
constexpr float EPS_ = 1e-5f;

__device__ __forceinline__ __bf16 f2bf(float f) { return (__bf16)f; }
__device__ __forceinline__ float  bf2f(__bf16 h) { return (float)h; }

// ---- CDNA5 async global->LDS copy (ASYNCcnt) -------------------------------
// Per-lane: LDS[lds_byte] = MEM[saddr + gl_off] (16 bytes). Generic pointers
// to LDS carry the LDS byte offset in their low 32 bits.
__device__ __forceinline__ unsigned lds_off(const void* p) {
  return (unsigned)(uintptr_t)p;
}
__device__ __forceinline__ void async_ld_b128(unsigned lds_byte, unsigned gl_off,
                                              const void* base) {
  asm volatile("global_load_async_to_lds_b128 %0, %1, %2 offset:0"
               :
               : "v"(lds_byte), "v"(gl_off), "s"(base)
               : "memory");
}
template <int N>
__device__ __forceinline__ void wait_asynccnt() {
  asm volatile("s_wait_asynccnt %0" : : "i"(N) : "memory");
}

// ---- WMMA operand gathers (documented wave32 VGPR layouts) -----------------
__device__ __forceinline__ v16bf gatherA(const __bf16* p, int lane) {
  const int off = (lane & 16) ? 8 : 0;
  v16bf a;
#pragma unroll
  for (int i = 0; i < 8; ++i) {
    const int kb = ((i >> 2) << 4) + off + ((i & 3) << 1);
    a[2 * i]     = p[kb];
    a[2 * i + 1] = p[kb + 1];
  }
  return a;
}
__device__ __forceinline__ v16bf gatherB(const __bf16* p, int lane) {
  const int off = (lane & 16) ? 8 : 0;
  v16bf b;
#pragma unroll
  for (int i = 0; i < 8; ++i) {
    const int kb = ((i >> 2) << 4) + off + ((i & 3) << 1);
    b[2 * i]     = p[kb];
    b[2 * i + 1] = p[kb + 1];
  }
  return b;
}
__device__ __forceinline__ v8f wmma_bf16(v16bf a, v16bf b, v8f c) {
  return __builtin_amdgcn_wmma_f32_16x16x32_bf16(
      false, a, false, b, (short)0, c, false, false);
}

// ---------------------------------------------------------------------------
__global__ __launch_bounds__(256) void mask_in_kernel(
    const float* __restrict__ x, const float* __restrict__ xm,
    float* __restrict__ y) {
  const size_t i = (size_t)blockIdx.x * 256 + threadIdx.x;
  const size_t total = (size_t)B_ * D_ * T_;
  if (i >= total) return;
  const int t = (int)(i % T_);
  const int b = (int)(i / ((size_t)D_ * T_));
  y[i] = x[i] * xm[b * T_ + t];
}

__global__ __launch_bounds__(256) void mask_out_kernel(
    const float* __restrict__ x, const float* __restrict__ xm,
    float* __restrict__ y) {
  const size_t i = (size_t)blockIdx.x * 256 + threadIdx.x;
  const size_t total = (size_t)B_ * D_ * T_;
  if (i >= total) return;
  const int t = (int)(i % T_);
  const int b = (int)(i / ((size_t)D_ * T_));
  y[i] = x[i] * xm[b * T_ + t];
}

// ---------------------------------------------------------------------------
// Generic bf16-WMMA GEMM over one batch slice: C[m, t] = sum_k W[m,k] * X(k, t)
enum { BM_X1X1 = 0, BM_HEADS = 1, BM_CONV3 = 2 };
enum { EM_QKV = 0, EM_QKV_T = 1, EM_RELU = 2, EM_RESID = 3 };

template <int BMODE, int EMODE>
__global__ __launch_bounds__(256) void gemm_kernel(
    const float* __restrict__ W,      // [M, Ktot] row-major
    const float* __restrict__ bias,   // [M]
    const float* __restrict__ X,      // per BMODE
    const float* __restrict__ res,    // residual [B,M,T] (EM_RESID)
    const float* __restrict__ xm,     // mask [B,T]
    void* __restrict__ Yv,
    int M, int Ktot, int Cin, float scale, int apply_mask_y) {
  __shared__ __bf16 As[64 * 40];     // [m][k], k contiguous (pad 40)
  __shared__ __bf16 Bs[64 * 40];     // transposed: [n][k], k contiguous
  const int b  = blockIdx.z;
  const int n0 = blockIdx.x * 64;
  const int m0 = blockIdx.y * 64;
  const int tid = threadIdx.x, wv = tid >> 5, lane = tid & 31;
  const int l15 = lane & 15, half = (lane >> 4) & 1;
  const int mi = wv & 3, ni0 = (wv >> 2) * 2;
  v8f c0 = {}, c1 = {};

  for (int k0 = 0; k0 < Ktot; k0 += 32) {
#pragma unroll
    for (int i = 0; i < 8; ++i) {
      const int e = tid + 256 * i;
      const int m = e >> 5, kk = e & 31;
      As[m * 40 + kk] = f2bf(W[(size_t)(m0 + m) * Ktot + k0 + kk]);
    }
#pragma unroll
    for (int i = 0; i < 8; ++i) {
      const int e = tid + 256 * i;
      const int kk = e >> 6, n = e & 63;
      const int kg = k0 + kk;
      const int t = n0 + n;
      float v;
      if constexpr (BMODE == BM_X1X1) {
        v = X[((size_t)b * Cin + kg) * T_ + t];
      } else if constexpr (BMODE == BM_HEADS) {
        const int h = kg / KC_, cc = kg % KC_;
        v = X[(((size_t)b * H_ + h) * T_ + t) * KC_ + cc];
      } else {  // BM_CONV3: fused K = c*KS + tap, masked input, pad=1
        const int cch = kg / KS_, tap = kg % KS_;
        const int tt = t + tap - 1;
        v = (tt >= 0 && tt < T_)
                ? X[((size_t)b * Cin + cch) * T_ + tt] * xm[b * T_ + tt]
                : 0.f;
      }
      Bs[n * 40 + kk] = f2bf(v);
    }
    __syncthreads();
    const v16bf a  = gatherA(&As[(mi * 16 + l15) * 40], lane);
    const v16bf b0 = gatherB(&Bs[(ni0 * 16 + l15) * 40], lane);
    const v16bf b1 = gatherB(&Bs[((ni0 + 1) * 16 + l15) * 40], lane);
    c0 = wmma_bf16(a, b0, c0);
    c1 = wmma_bf16(a, b1, c1);
    __syncthreads();
  }

#pragma unroll
  for (int p = 0; p < 2; ++p) {
    const v8f cc = p ? c1 : c0;
    const int ni = ni0 + p;
#pragma unroll
    for (int j = 0; j < 8; ++j) {
      const int m = m0 + mi * 16 + j + 8 * half;
      const int t = n0 + ni * 16 + l15;
      float v = cc[j] + bias[m];
      if constexpr (EMODE == EM_QKV) {          // bf16 [bh, t, c]
        v *= scale;
        const int h = m / KC_, ch = m % KC_;
        ((__bf16*)Yv)[(((size_t)b * H_ + h) * T_ + t) * KC_ + ch] = f2bf(v);
      } else if constexpr (EMODE == EM_QKV_T) { // bf16 [bh, c, t] (V)
        const int h = m / KC_, ch = m % KC_;
        ((__bf16*)Yv)[(((size_t)b * H_ + h) * KC_ + ch) * T_ + t] = f2bf(v);
      } else if constexpr (EMODE == EM_RELU) {
        ((float*)Yv)[((size_t)b * M + m) * T_ + t] = fmaxf(v, 0.f);
      } else {  // EM_RESID
        if (apply_mask_y) v *= xm[b * T_ + t];
        ((float*)Yv)[((size_t)b * M + m) * T_ + t] =
            v + res[((size_t)b * M + m) * T_ + t];
      }
    }
  }
}

// ---------------------------------------------------------------------------
// relK[bh, t, d] = sum_c qs[bh,t,c] * emb_k[d, c]   (d = delta + WS, 21 wide)
__global__ __launch_bounds__(256) void relk_kernel(
    const __bf16* __restrict__ q, const float* __restrict__ embk,
    float* __restrict__ rk) {
  const int idx = blockIdx.x * 256 + threadIdx.x;
  const int total = B_ * H_ * T_ * NR_;
  if (idx >= total) return;
  const int d = idx % NR_;
  const int bt = idx / NR_;  // bh*T + t
  const __bf16* qp = q + (size_t)bt * KC_;
  const float* ep = embk + d * KC_;
  float s = 0.f;
#pragma unroll 8
  for (int c = 0; c < KC_; ++c) s += bf2f(qp[c]) * ep[c];
  rk[idx] = s;
}

// ---------------------------------------------------------------------------
// Flash attention, banded relative-position terms, async double-buffered K/V.
// Block: 128 threads (4 waves); each wave owns 16 q-rows (block covers 64).
__global__ __launch_bounds__(128) void attn_kernel(
    const __bf16* __restrict__ q, const __bf16* __restrict__ k,
    const __bf16* __restrict__ vT, const float* __restrict__ rk,
    const float* __restrict__ embv, const float* __restrict__ xm,
    float* __restrict__ out) {
  __shared__ __align__(16) __bf16 Qs[64 * 96];      // [t_local][c]
  __shared__ __align__(16) __bf16 Kt[2][32 * 96];   // [s_local][c]
  __shared__ __align__(16) __bf16 Vt[2][96 * 32];   // [d][s_local]
  __shared__ __align__(16) __bf16 Et[96 * 32];      // [d][delta] (zero-padded)
  __shared__ __align__(16) __bf16 Ps[4][16 * 32];   // per-wave A staging

  const int bh = blockIdx.y, b = bh / H_;
  const int t0 = blockIdx.x * 64;
  const int tid = threadIdx.x, wv = tid >> 5, lane = tid & 31;
  const int l15 = lane & 15, half = (lane >> 4) & 1;
  const size_t base = (size_t)bh * T_ * KC_;
  constexpr int NIT = T_ / 32;

  // Async stage Q tile (contiguous 12KB): 6 b128 per wave.
  {
    const void* gq = (const void*)(q + base + (size_t)t0 * KC_);
    const unsigned l0 = lds_off(&Qs[0]);
#pragma unroll
    for (int i = 0; i < 6; ++i) {
      const unsigned e = (unsigned)tid + 128u * i;
      async_ld_b128(l0 + e * 16, e * 16, gq);
    }
  }
  // Et: fp32 -> bf16 with zero pad (deltas 21..31), plain stores.
  for (int e = tid; e < 96 * 32; e += 128) {
    const int d = e >> 5, dl = e & 31;
    Et[e] = f2bf(dl < NR_ ? embv[dl * 96 + d] : 0.f);
  }

  // K chunk: contiguous 6KB; V chunk: 96 rows x 64B from [bh, d, t] layout.
  auto issue_chunk = [&](int it) {
    const int s0 = it * 32;
    const int bi = it & 1;
    const void* gk = (const void*)(k + base + (size_t)s0 * KC_);
    const unsigned lk = lds_off(&Kt[bi][0]);
#pragma unroll
    for (int i = 0; i < 3; ++i) {
      const unsigned e = (unsigned)tid + 128u * i;
      async_ld_b128(lk + e * 16, e * 16, gk);
    }
    const void* gv = (const void*)(vT + (size_t)bh * KC_ * T_);
    const unsigned lv = lds_off(&Vt[bi][0]);
#pragma unroll
    for (int i = 0; i < 3; ++i) {
      const int e = tid + 128 * i;          // 0..383
      const int d = e >> 2, g = e & 3;
      async_ld_b128(lv + (unsigned)(d * 32 + g * 8) * 2,
                    (unsigned)((d * T_ + s0 + g * 8) * 2), gv);
    }
  };
  issue_chunk(0);
  issue_chunk(1);

  v8f oacc[6];
#pragma unroll
  for (int i = 0; i < 6; ++i) oacc[i] = (v8f){};
  v8f wb0 = {}, wb1 = {};
  float mrow[8], lrow[8], qmv[8];
  int trow[8];
#pragma unroll
  for (int j = 0; j < 8; ++j) {
    mrow[j] = -3.0e38f;
    lrow[j] = 0.f;
    trow[j] = t0 + wv * 16 + j + 8 * half;
    qmv[j]  = xm[b * T_ + trow[j]];
  }
  const int tlo = t0 + wv * 16, thi = tlo + 15;

  for (int it = 0; it < NIT; ++it) {
    const int s0 = it * 32;
    // Drain everything except the single in-flight next chunk (6 ops/wave).
    if (it + 1 < NIT) wait_asynccnt<6>(); else wait_asynccnt<0>();
    __syncthreads();
    const __bf16* KtB = &Kt[it & 1][0];
    const __bf16* VtB = &Vt[it & 1][0];

    // S = Q * K^T  (two 16x16 tiles over 32 keys; K-dim = KC = 96)
    v8f S0 = {}, S1 = {};
    const __bf16* qrow = &Qs[(wv * 16 + l15) * 96];
#pragma unroll
    for (int c0 = 0; c0 < 96; c0 += 32) {
      const v16bf a  = gatherA(qrow + c0, lane);
      const v16bf b0 = gatherB(&KtB[l15 * 96 + c0], lane);
      const v16bf b1 = gatherB(&KtB[(16 + l15) * 96 + c0], lane);
      S0 = wmma_bf16(a, b0, S0);
      S1 = wmma_bf16(a, b1, S1);
    }

    const int sc0 = s0 + l15, sc1 = s0 + 16 + l15;
    const float km0 = xm[b * T_ + sc0], km1 = xm[b * T_ + sc1];
    const bool band = (s0 + 31 >= tlo - WS_) && (s0 <= thi + WS_);
#pragma unroll
    for (int j = 0; j < 8; ++j) {
      float a0 = S0[j], a1 = S1[j];
      if (band) {
        const int d0 = sc0 - trow[j];
        if (d0 >= -WS_ && d0 <= WS_)
          a0 += rk[((size_t)bh * T_ + trow[j]) * NR_ + d0 + WS_];
        const int d1 = sc1 - trow[j];
        if (d1 >= -WS_ && d1 <= WS_)
          a1 += rk[((size_t)bh * T_ + trow[j]) * NR_ + d1 + WS_];
      }
      if (qmv[j] * km0 == 0.f) a0 = -1e4f;
      if (qmv[j] * km1 == 0.f) a1 = -1e4f;
      S0[j] = a0;
      S1[j] = a1;
    }

    // Online softmax: row max/sum across the 16 lanes owning each row.
    float corr[8];
#pragma unroll
    for (int j = 0; j < 8; ++j) {
      float mx = fmaxf(S0[j], S1[j]);
#pragma unroll
      for (int o = 1; o < 16; o <<= 1) mx = fmaxf(mx, __shfl_xor(mx, o, 32));
      const float nm = fmaxf(mrow[j], mx);
      corr[j] = __expf(mrow[j] - nm);
      mrow[j] = nm;
      S0[j] = __expf(S0[j] - nm);
      S1[j] = __expf(S1[j] - nm);
      float rs = S0[j] + S1[j];
#pragma unroll
      for (int o = 1; o < 16; o <<= 1) rs += __shfl_xor(rs, o, 32);
      lrow[j] = lrow[j] * corr[j] + rs;
    }
#pragma unroll
    for (int d = 0; d < 6; ++d)
#pragma unroll
      for (int j = 0; j < 8; ++j) oacc[d][j] *= corr[j];
#pragma unroll
    for (int j = 0; j < 8; ++j) {
      wb0[j] *= corr[j];
      wb1[j] *= corr[j];
    }

    // Stage P (16x32) per wave; reuse as WMMA A operand.
    __bf16* pp = &Ps[wv][0];
#pragma unroll
    for (int j = 0; j < 8; ++j) {
      const int r = j + 8 * half;
      pp[r * 32 + l15]      = f2bf(S0[j]);
      pp[r * 32 + 16 + l15] = f2bf(S1[j]);
    }
    const v16bf pa = gatherA(&Ps[wv][l15 * 32], lane);
#pragma unroll
    for (int d = 0; d < 6; ++d) {
      const v16bf bv = gatherB(&VtB[(d * 16 + l15) * 32], lane);
      oacc[d] = wmma_bf16(pa, bv, oacc[d]);
    }

    // Band accumulator w[t, delta] += P[t, t+delta-WS].
    if (band) {
#pragma unroll
      for (int j = 0; j < 8; ++j) {
        const int r = j + 8 * half;
        const int sl0 = trow[j] + (l15 - WS_) - s0;
        if (sl0 >= 0 && sl0 < 32) wb0[j] += bf2f(pp[r * 32 + sl0]);
        if (16 + l15 <= 2 * WS_) {
          const int sl1 = trow[j] + (16 + l15 - WS_) - s0;
          if (sl1 >= 0 && sl1 < 32) wb1[j] += bf2f(pp[r * 32 + sl1]);
        }
      }
    }
    __syncthreads();
    if (it + 2 < NIT) issue_chunk(it + 2);
  }

  // out += w[t, :] * emb_v  (K = 32, deltas 21..31 zero-padded)
  __bf16* pp = &Ps[wv][0];
#pragma unroll
  for (int j = 0; j < 8; ++j) {
    const int r = j + 8 * half;
    pp[r * 32 + l15]      = f2bf(wb0[j]);
    pp[r * 32 + 16 + l15] = f2bf(wb1[j]);
  }
  const v16bf wa = gatherA(&Ps[wv][l15 * 32], lane);
#pragma unroll
  for (int d = 0; d < 6; ++d) {
    const v16bf be = gatherB(&Et[(d * 16 + l15) * 32], lane);
    oacc[d] = wmma_bf16(wa, be, oacc[d]);
  }

  float inv[8];
#pragma unroll
  for (int j = 0; j < 8; ++j) inv[j] = (lrow[j] > 0.f) ? 1.f / lrow[j] : 0.f;
#pragma unroll
  for (int d = 0; d < 6; ++d)
#pragma unroll
    for (int j = 0; j < 8; ++j)
      out[base + (size_t)trow[j] * 96 + d * 16 + l15] = oacc[d][j] * inv[j];
}

// ---------------------------------------------------------------------------
// Channels-first LayerNorm over D=192 per (b,t). One warp per column.
__global__ __launch_bounds__(256) void ln_kernel(
    const float* __restrict__ in, const float* __restrict__ g,
    const float* __restrict__ be, float* __restrict__ out) {
  const int col = blockIdx.x * 8 + (threadIdx.x >> 5);  // b*T + t
  const int lane = threadIdx.x & 31;
  const int b = col / T_, t = col % T_;
  const float* cp = in + (size_t)b * D_ * T_ + t;
  float vals[6];
  float s = 0.f;
#pragma unroll
  for (int i = 0; i < 6; ++i) {
    vals[i] = cp[(size_t)(lane + 32 * i) * T_];
    s += vals[i];
  }
#pragma unroll
  for (int o = 1; o < 32; o <<= 1) s += __shfl_xor(s, o, 32);
  const float mean = s / D_;
  float vs = 0.f;
#pragma unroll
  for (int i = 0; i < 6; ++i) {
    const float d = vals[i] - mean;
    vs += d * d;
  }
#pragma unroll
  for (int o = 1; o < 32; o <<= 1) vs += __shfl_xor(vs, o, 32);
  const float r = rsqrtf(vs / D_ + EPS_);
#pragma unroll
  for (int i = 0; i < 6; ++i) {
    const int c = lane + 32 * i;
    out[(size_t)b * D_ * T_ + (size_t)c * T_ + t] =
        (vals[i] - mean) * r * g[c] + be[c];
  }
}

// ---------------------------------------------------------------------------
extern "C" void kernel_launch(void* const* d_in, const int* in_sizes, int n_in,
                              void* d_out, int out_size, void* d_ws,
                              size_t ws_size, hipStream_t stream) {
  const float* x     = (const float*)d_in[0];
  const float* xm    = (const float*)d_in[1];
  const float* Wq    = (const float*)d_in[2];
  const float* bq    = (const float*)d_in[3];
  const float* Wk    = (const float*)d_in[4];
  const float* bk    = (const float*)d_in[5];
  const float* Wv    = (const float*)d_in[6];
  const float* bv    = (const float*)d_in[7];
  const float* Wo    = (const float*)d_in[8];
  const float* bo    = (const float*)d_in[9];
  const float* emb_k = (const float*)d_in[10];
  const float* emb_v = (const float*)d_in[11];
  const float* g1    = (const float*)d_in[12];
  const float* b1    = (const float*)d_in[13];
  const float* Wf1   = (const float*)d_in[14];
  const float* bf1   = (const float*)d_in[15];
  const float* Wf2   = (const float*)d_in[16];
  const float* bf2   = (const float*)d_in[17];
  const float* g2    = (const float*)d_in[18];
  const float* b2    = (const float*)d_in[19];

  const size_t NBDT = (size_t)B_ * D_ * T_;     // 1.57M elements
  char* w = (char*)d_ws;                        // ~55 MB total
  float* xbuf = (float*)w;  w += NBDT * 4;
  float* tmp  = (float*)w;  w += NBDT * 4;
  float* ob   = (float*)w;  w += NBDT * 4;
  float* rk   = (float*)w;  w += ((size_t)B_ * H_ * T_ * NR_ * 4 + 255) & ~(size_t)255;
  float* ff   = (float*)w;  w += (size_t)B_ * FF_ * T_ * 4;
  __bf16* qb  = (__bf16*)w; w += NBDT * 2;
  __bf16* kb  = (__bf16*)w; w += NBDT * 2;
  __bf16* vT  = (__bf16*)w; w += NBDT * 2;

  const dim3 blk256(256), blk128(128);
  const dim3 gMaskX((unsigned)((NBDT + 255) / 256));
  const dim3 gGemmD(T_ / 64, D_ / 64, B_);      // M = 192
  const dim3 gGemmF(T_ / 64, FF_ / 64, B_);     // M = 768
  const dim3 gRelK((B_ * H_ * T_ * NR_ + 255) / 256);
  const dim3 gAttn(T_ / 64, B_ * H_);
  const dim3 gLN(B_ * T_ / 8);
  const float qscale = 1.0f / sqrtf((float)KC_);

  mask_in_kernel<<<gMaskX, blk256, 0, stream>>>(x, xm, xbuf);

  for (int l = 0; l < L_; ++l) {
    const float* Wq_l = Wq + (size_t)l * D_ * D_;
    const float* Wk_l = Wk + (size_t)l * D_ * D_;
    const float* Wv_l = Wv + (size_t)l * D_ * D_;
    const float* Wo_l = Wo + (size_t)l * D_ * D_;
    const float* bq_l = bq + (size_t)l * D_;
    const float* bk_l = bk + (size_t)l * D_;
    const float* bv_l = bv + (size_t)l * D_;
    const float* bo_l = bo + (size_t)l * D_;
    const float* ek_l = emb_k + (size_t)l * NR_ * KC_;
    const float* ev_l = emb_v + (size_t)l * NR_ * KC_;
    const float* g1_l = g1 + (size_t)l * D_;
    const float* b1_l = b1 + (size_t)l * D_;
    const float* g2_l = g2 + (size_t)l * D_;
    const float* b2_l = b2 + (size_t)l * D_;
    const float* Wf1_l = Wf1 + (size_t)l * FF_ * D_ * KS_;
    const float* bf1_l = bf1 + (size_t)l * FF_;
    const float* Wf2_l = Wf2 + (size_t)l * D_ * FF_ * KS_;
    const float* bf2_l = bf2 + (size_t)l * D_;

    // Q (pre-scaled) / K as bf16 [bh,t,c]; V as bf16 [bh,c,t] for async tiles.
    gemm_kernel<BM_X1X1, EM_QKV><<<gGemmD, blk256, 0, stream>>>(
        Wq_l, bq_l, xbuf, xbuf, xm, qb, D_, D_, D_, qscale, 0);
    gemm_kernel<BM_X1X1, EM_QKV><<<gGemmD, blk256, 0, stream>>>(
        Wk_l, bk_l, xbuf, xbuf, xm, kb, D_, D_, D_, 1.0f, 0);
    gemm_kernel<BM_X1X1, EM_QKV_T><<<gGemmD, blk256, 0, stream>>>(
        Wv_l, bv_l, xbuf, xbuf, xm, vT, D_, D_, D_, 1.0f, 0);

    // Banded relative-K logits, then flash attention.
    relk_kernel<<<gRelK, blk256, 0, stream>>>(qb, ek_l, rk);
    attn_kernel<<<gAttn, blk128, 0, stream>>>(qb, kb, vT, rk, ev_l, xm, ob);

    // O-projection + residual, LayerNorm 1.
    gemm_kernel<BM_HEADS, EM_RESID><<<gGemmD, blk256, 0, stream>>>(
        Wo_l, bo_l, ob, xbuf, xm, tmp, D_, D_, D_, 1.0f, 0);
    ln_kernel<<<gLN, blk256, 0, stream>>>(tmp, g1_l, b1_l, xbuf);

    // FFN: 3-tap conv as fused-K GEMMs; relu; residual; LayerNorm 2.
    gemm_kernel<BM_CONV3, EM_RELU><<<gGemmF, blk256, 0, stream>>>(
        Wf1_l, bf1_l, xbuf, xbuf, xm, ff, FF_, D_ * KS_, D_, 1.0f, 0);
    gemm_kernel<BM_CONV3, EM_RESID><<<gGemmD, blk256, 0, stream>>>(
        Wf2_l, bf2_l, ff, xbuf, xm, tmp, D_, FF_ * KS_, FF_, 1.0f, 1);
    ln_kernel<<<gLN, blk256, 0, stream>>>(tmp, g2_l, b2_l, xbuf);
  }

  mask_out_kernel<<<gMaskX, blk256, 0, stream>>>(xbuf, xm, (float*)d_out);
  (void)in_sizes; (void)n_in; (void)out_size; (void)ws_size;
}
